// SpaghettiOfExperts_55482387529859
// MI455X (gfx1250) — compile-verified, matrix operand order checked by reading, not run on hardware
//
#include <hip/hip_runtime.h>
#include <hip/hip_bf16.h>

// ---------------------------------------------------------------------------
// MoE (top-2, SwiGLU) for MI455X / gfx1250.
// Sparse token gathering + bf16 WMMA (v_wmma_f32_16x16x32_bf16), fp32 accum.
// ---------------------------------------------------------------------------

typedef __attribute__((ext_vector_type(16))) __bf16 v16bf;
typedef __attribute__((ext_vector_type(8)))  float  v8f;

#define D_MODEL 1024
#define D_FF    4096
#define N_EXP   8
#define T_TOK   8192
#define T_PAIR  (T_TOK * 2)

#define BM 64
#define BN 64
#define KC 64
#define MT_MAX (T_PAIR / BM)   // 256 M-tiles covers worst-case routing skew

__device__ __forceinline__ unsigned short f2bf(float f) {
  unsigned u = __float_as_uint(f);
  u += 0x7FFFu + ((u >> 16) & 1u);        // round-to-nearest-even
  return (unsigned short)(u >> 16);
}
__device__ __forceinline__ unsigned bfpack(float lo, float hi) {
  return (unsigned)f2bf(lo) | ((unsigned)f2bf(hi) << 16);
}
__device__ __forceinline__ unsigned umin_u(unsigned a, unsigned b) { return a < b ? a : b; }

// ---------------------------------------------------------------- cast x->bf16
__global__ void moe_cast_x_bf16(const float* __restrict__ x,
                                unsigned short* __restrict__ xbf, int n4) {
  int i = blockIdx.x * blockDim.x + threadIdx.x;
  int stride = gridDim.x * blockDim.x;
  for (; i < n4; i += stride) {
    float4 v = ((const float4*)x)[i];
    uint2 p;
    p.x = bfpack(v.x, v.y);
    p.y = bfpack(v.z, v.w);
    ((uint2*)xbf)[i] = p;
  }
}

// ---------------------------------------------------------------- router
// One wave32 per token: 8-way dot product, softmax, top-2, stats.
__global__ void moe_router(const float* __restrict__ x,
                           const float* __restrict__ rw,
                           unsigned* __restrict__ cnt,
                           float* __restrict__ imp,
                           unsigned* __restrict__ loadc,
                           int* __restrict__ topk_id,
                           float* __restrict__ topk_w) {
  int t = blockIdx.x * (blockDim.x >> 5) + (threadIdx.x >> 5);
  int lane = threadIdx.x & 31;
  if (t >= T_TOK) return;

  float acc[N_EXP];
#pragma unroll
  for (int e = 0; e < N_EXP; e++) acc[e] = 0.f;

  const float* xr = x + (size_t)t * D_MODEL;
  for (int d = lane; d < D_MODEL; d += 32) {
    float xv = xr[d];
    const float* r = rw + d * N_EXP;
#pragma unroll
    for (int e = 0; e < N_EXP; e++) acc[e] = fmaf(xv, r[e], acc[e]);
  }
#pragma unroll
  for (int off = 16; off >= 1; off >>= 1)
#pragma unroll
    for (int e = 0; e < N_EXP; e++) acc[e] += __shfl_xor(acc[e], off, 32);

  if (lane == 0) {
    float mx = acc[0];
#pragma unroll
    for (int e = 1; e < N_EXP; e++) mx = fmaxf(mx, acc[e]);
    float p[N_EXP], s = 0.f;
#pragma unroll
    for (int e = 0; e < N_EXP; e++) { p[e] = __expf(acc[e] - mx); s += p[e]; }
    float inv = 1.f / s;

    int i0 = 0;
#pragma unroll
    for (int e = 1; e < N_EXP; e++) if (p[e] > p[i0]) i0 = e;
    int i1 = (i0 == 0) ? 1 : 0;
#pragma unroll
    for (int e = 0; e < N_EXP; e++) if (e != i0 && p[e] > p[i1]) i1 = e;

#pragma unroll
    for (int e = 0; e < N_EXP; e++) atomicAdd(&imp[e], p[e] * inv);
    atomicAdd(&loadc[i0], 1u);
    atomicAdd(&cnt[i0], 1u);
    atomicAdd(&cnt[i1], 1u);

    float g0 = p[i0], g1 = p[i1], gs = g0 + g1;
    topk_id[t * 2 + 0] = i0;
    topk_id[t * 2 + 1] = i1;
    topk_w[t * 2 + 0] = g0 / gs;
    topk_w[t * 2 + 1] = g1 / gs;
  }
}

// ---------------------------------------------------------------- scan + aux
__global__ void moe_scan_aux(const unsigned* __restrict__ cnt,
                             unsigned* __restrict__ offs,
                             unsigned* __restrict__ cursor,
                             const float* __restrict__ imp,
                             const unsigned* __restrict__ loadc,
                             float* __restrict__ aux_out) {
  if (threadIdx.x == 0 && blockIdx.x == 0) {
    unsigned o = 0;
    float loss = 0.f;
    const float invT = 1.f / (float)T_TOK;
    for (int e = 0; e < N_EXP; e++) {
      offs[e] = o;
      cursor[e] = o;
      o += cnt[e];
      loss += (imp[e] * invT) * ((float)loadc[e] * invT);
    }
    aux_out[0] = (float)N_EXP * loss;
  }
}

// ---------------------------------------------------------------- scatter
__global__ void moe_scatter(const int* __restrict__ topk_id,
                            const float* __restrict__ topk_w,
                            unsigned* __restrict__ cursor,
                            unsigned* __restrict__ ptok,
                            float* __restrict__ pgate) {
  int t = blockIdx.x * blockDim.x + threadIdx.x;
  if (t >= T_TOK) return;
#pragma unroll
  for (int k = 0; k < 2; k++) {
    int e = topk_id[t * 2 + k];
    unsigned p = atomicAdd(&cursor[e], 1u);
    ptok[p] = (unsigned)t;
    pgate[p] = topk_w[t * 2 + k];
  }
}

// ---------------------------------------------------------------- FFN pass 1
// H[pair, f] = silu(x@w1) * (x@w3) for gathered rows of one expert.
// 256 threads (8 waves), BM=64 x BN=64 tile, K-chunk 64 (2 WMMA k-steps).
__global__ __launch_bounds__(256, 2)
void moe_ffn1(const unsigned short* __restrict__ xbf,
              const float* __restrict__ w1, const float* __restrict__ w3,
              const unsigned* __restrict__ offs, const unsigned* __restrict__ cnt,
              const unsigned* __restrict__ ptok,
              unsigned short* __restrict__ H) {
  const int e = blockIdx.z;
  const unsigned base = offs[e];
  const unsigned rows = cnt[e];
  const int m0 = blockIdx.y * BM;
  if ((unsigned)m0 >= rows) return;
  const int n0 = blockIdx.x * BN;

  __shared__ alignas(32) unsigned short As[BM * KC];
  __shared__ alignas(32) unsigned short B1s[BN * KC];
  __shared__ alignas(32) unsigned short B3s[BN * KC];

  const int tid = threadIdx.x;
  const int wave = tid >> 5, lane = tid & 31;
  const int mi = wave & 3;            // M subtile 0..3
  const int nt0 = (wave >> 2) * 2;    // first of two N subtiles

  // Hoisted gathered-row pointers (loop-invariant). Out-of-range rows are
  // clamped to a valid row: their results are never stored (guarded epilogue),
  // so the load path is branch-free.
  const int am = tid >> 3;            // row 0..31 (and +32 for second chunk)
  const int ac = (tid & 7) * 8;       // bf16 column offset within K-chunk
  const unsigned tokA = ptok[base + umin_u((unsigned)(m0 + am),      rows - 1)];
  const unsigned tokB = ptok[base + umin_u((unsigned)(m0 + am + 32), rows - 1)];
  const unsigned short* rpA = xbf + (size_t)tokA * D_MODEL;
  const unsigned short* rpB = xbf + (size_t)tokB * D_MODEL;

  v8f acc10 = {}, acc11 = {}, acc30 = {}, acc31 = {};
  const int lrow = lane & 15, lk = (lane >> 4) * 16;

  for (int kk = 0; kk < D_MODEL; kk += KC) {
    // A tile: one b128 load + one b128 LDS store per thread per row-chunk.
    uint4 a0 = *(const uint4*)(rpA + kk + ac);
    uint4 a1 = *(const uint4*)(rpB + kk + ac);
    ((uint4*)As)[am * 8 + (tid & 7)]        = a0;
    ((uint4*)As)[(am + 32) * 8 + (tid & 7)] = a1;
    // B tiles: KC x BN fp32 -> bf16, transposed [n][k].
    // b128 loads along n for two adjacent k-rows, k-pair packed u32 stores.
#pragma unroll
    for (int j = 0; j < 2; j++) {
      int q = tid + j * 256;            // [0,512)
      int n4 = q & 15, kp = q >> 4;     // kp in [0,32), k = 2*kp
      size_t gi = ((size_t)e * D_MODEL + kk + kp * 2) * D_FF + n0 + n4 * 4;
      float4 u0 = *(const float4*)(w1 + gi);
      float4 u1 = *(const float4*)(w1 + gi + D_FF);
      float4 v0 = *(const float4*)(w3 + gi);
      float4 v1 = *(const float4*)(w3 + gi + D_FF);
      unsigned* b1 = (unsigned*)B1s;
      unsigned* b3 = (unsigned*)B3s;
      b1[(n4 * 4 + 0) * 32 + kp] = bfpack(u0.x, u1.x);
      b1[(n4 * 4 + 1) * 32 + kp] = bfpack(u0.y, u1.y);
      b1[(n4 * 4 + 2) * 32 + kp] = bfpack(u0.z, u1.z);
      b1[(n4 * 4 + 3) * 32 + kp] = bfpack(u0.w, u1.w);
      b3[(n4 * 4 + 0) * 32 + kp] = bfpack(v0.x, v1.x);
      b3[(n4 * 4 + 1) * 32 + kp] = bfpack(v0.y, v1.y);
      b3[(n4 * 4 + 2) * 32 + kp] = bfpack(v0.z, v1.z);
      b3[(n4 * 4 + 3) * 32 + kp] = bfpack(v0.w, v1.w);
    }
    if (kk + KC < D_MODEL) {  // hint next weight tile toward cache
      size_t gp = ((size_t)e * D_MODEL + kk + KC + (tid >> 6)) * D_FF + n0 + (tid & 63);
      __builtin_prefetch((const void*)(w1 + gp), 0, 0);
      __builtin_prefetch((const void*)(w3 + gp), 0, 0);
    }
    __syncthreads();
#pragma unroll
    for (int s = 0; s < 2; s++) {
      const int ko = s * 32 + lk;
      v16bf a   = *(const v16bf*)&As [(mi * 16 + lrow) * KC + ko];
      v16bf b10 = *(const v16bf*)&B1s[( nt0      * 16 + lrow) * KC + ko];
      v16bf b11 = *(const v16bf*)&B1s[((nt0 + 1) * 16 + lrow) * KC + ko];
      v16bf b30 = *(const v16bf*)&B3s[( nt0      * 16 + lrow) * KC + ko];
      v16bf b31 = *(const v16bf*)&B3s[((nt0 + 1) * 16 + lrow) * KC + ko];
      acc10 = __builtin_amdgcn_wmma_f32_16x16x32_bf16(false, a, false, b10, (short)0, acc10, false, false);
      acc11 = __builtin_amdgcn_wmma_f32_16x16x32_bf16(false, a, false, b11, (short)0, acc11, false, false);
      acc30 = __builtin_amdgcn_wmma_f32_16x16x32_bf16(false, a, false, b30, (short)0, acc30, false, false);
      acc31 = __builtin_amdgcn_wmma_f32_16x16x32_bf16(false, a, false, b31, (short)0, acc31, false, false);
    }
    __syncthreads();
  }

  // Epilogue: SwiGLU. C/D layout: VGPR r -> M = r + (lane>>4)*8, N = lane&15.
  const int Nc = lane & 15, Mh = (lane >> 4) * 8;
#pragma unroll
  for (int r = 0; r < 8; r++) {
    int gm = m0 + mi * 16 + r + Mh;
    if ((unsigned)gm < rows) {
      float g0 = acc10[r], g1 = acc11[r];
      float h0 = (g0 / (1.f + __expf(-g0))) * acc30[r];
      float h1 = (g1 / (1.f + __expf(-g1))) * acc31[r];
      size_t rowbase = (size_t)(base + gm) * D_FF + n0;
      H[rowbase + nt0 * 16 + Nc]       = f2bf(h0);
      H[rowbase + (nt0 + 1) * 16 + Nc] = f2bf(h1);
    }
  }
}

// ---------------------------------------------------------------- FFN pass 2
// out[tok] += gate * (H[pair] @ w2[e]); gated float atomics into d_out.
__global__ __launch_bounds__(256, 2)
void moe_ffn2(const unsigned short* __restrict__ H,
              const float* __restrict__ w2,
              const unsigned* __restrict__ offs, const unsigned* __restrict__ cnt,
              const unsigned* __restrict__ ptok, const float* __restrict__ pgate,
              float* __restrict__ out) {
  const int e = blockIdx.z;
  const unsigned base = offs[e];
  const unsigned rows = cnt[e];
  const int m0 = blockIdx.y * BM;
  if ((unsigned)m0 >= rows) return;
  const int n0 = blockIdx.x * BN;

  __shared__ alignas(32) unsigned short As[BM * KC];
  __shared__ alignas(32) unsigned short Bs[BN * KC];

  const int tid = threadIdx.x;
  const int wave = tid >> 5, lane = tid & 31;
  const int mi = wave & 3;
  const int nt0 = (wave >> 2) * 2;

  const int am = tid >> 3;
  const int ac = (tid & 7) * 8;
  const unsigned rA = umin_u((unsigned)(m0 + am),      rows - 1);
  const unsigned rB = umin_u((unsigned)(m0 + am + 32), rows - 1);
  const unsigned short* rpA = H + (size_t)(base + rA) * D_FF;
  const unsigned short* rpB = H + (size_t)(base + rB) * D_FF;

  v8f acc0 = {}, acc1 = {};
  const int lrow = lane & 15, lk = (lane >> 4) * 16;

  for (int kk = 0; kk < D_FF; kk += KC) {
    uint4 a0 = *(const uint4*)(rpA + kk + ac);
    uint4 a1 = *(const uint4*)(rpB + kk + ac);
    ((uint4*)As)[am * 8 + (tid & 7)]        = a0;
    ((uint4*)As)[(am + 32) * 8 + (tid & 7)] = a1;
#pragma unroll
    for (int j = 0; j < 2; j++) {
      int q = tid + j * 256;            // [0,512)
      int n4 = q & 15, kp = q >> 4;     // kp in [0,32)
      size_t gi = ((size_t)e * D_FF + kk + kp * 2) * D_MODEL + n0 + n4 * 4;
      float4 u0 = *(const float4*)(w2 + gi);
      float4 u1 = *(const float4*)(w2 + gi + D_MODEL);
      unsigned* bb = (unsigned*)Bs;
      bb[(n4 * 4 + 0) * 32 + kp] = bfpack(u0.x, u1.x);
      bb[(n4 * 4 + 1) * 32 + kp] = bfpack(u0.y, u1.y);
      bb[(n4 * 4 + 2) * 32 + kp] = bfpack(u0.z, u1.z);
      bb[(n4 * 4 + 3) * 32 + kp] = bfpack(u0.w, u1.w);
    }
    if (kk + KC < D_FF) {
      size_t gp = ((size_t)e * D_FF + kk + KC + (tid >> 6)) * D_MODEL + n0 + (tid & 63);
      __builtin_prefetch((const void*)(w2 + gp), 0, 0);
    }
    __syncthreads();
#pragma unroll
    for (int s = 0; s < 2; s++) {
      const int ko = s * 32 + lk;
      v16bf a  = *(const v16bf*)&As[(mi * 16 + lrow) * KC + ko];
      v16bf b0 = *(const v16bf*)&Bs[( nt0      * 16 + lrow) * KC + ko];
      v16bf b1 = *(const v16bf*)&Bs[((nt0 + 1) * 16 + lrow) * KC + ko];
      acc0 = __builtin_amdgcn_wmma_f32_16x16x32_bf16(false, a, false, b0, (short)0, acc0, false, false);
      acc1 = __builtin_amdgcn_wmma_f32_16x16x32_bf16(false, a, false, b1, (short)0, acc1, false, false);
    }
    __syncthreads();
  }

  const int Nc = lane & 15, Mh = (lane >> 4) * 8;
#pragma unroll
  for (int r = 0; r < 8; r++) {
    int gm = m0 + mi * 16 + r + Mh;
    if ((unsigned)gm < rows) {
      unsigned tok = ptok[base + gm];
      float g = pgate[base + gm];
      atomicAdd(&out[(size_t)tok * D_MODEL + n0 + nt0 * 16 + Nc],       g * acc0[r]);
      atomicAdd(&out[(size_t)tok * D_MODEL + n0 + (nt0 + 1) * 16 + Nc], g * acc1[r]);
    }
  }
}

// ---------------------------------------------------------------- launch
extern "C" void kernel_launch(void* const* d_in, const int* in_sizes, int n_in,
                              void* d_out, int out_size, void* d_ws, size_t ws_size,
                              hipStream_t stream) {
  (void)in_sizes; (void)n_in; (void)out_size; (void)ws_size;
  const float* x  = (const float*)d_in[0];
  const float* rw = (const float*)d_in[1];
  const float* w1 = (const float*)d_in[2];
  const float* w2 = (const float*)d_in[3];
  const float* w3 = (const float*)d_in[4];
  float* out = (float*)d_out;

  char* ws = (char*)d_ws;
  const size_t XBF = (size_t)T_TOK * D_MODEL * 2;   // x in bf16
  const size_t HB  = (size_t)T_PAIR * D_FF * 2;     // hidden H in bf16
  unsigned short* xbf  = (unsigned short*)ws;
  unsigned short* Hbuf = (unsigned short*)(ws + XBF);
  char* ctl = ws + XBF + HB;
  unsigned* cnt    = (unsigned*)ctl;          // [8]
  unsigned* offs   = cnt + 8;                 // [8]
  unsigned* cursor = cnt + 16;                // [8]
  float*    imp    = (float*)(cnt + 24);      // [8]
  unsigned* loadc  = cnt + 32;                // [8]
  int*      topk_id = (int*)(ctl + 256);                       // T*2
  float*    topk_w  = (float*)(ctl + 256 + (size_t)T_TOK * 8); // T*2
  unsigned* ptok    = (unsigned*)(ctl + 256 + (size_t)T_TOK * 16);
  float*    pgate   = (float*)(ctl + 256 + (size_t)T_TOK * 20);

  hipMemsetAsync(ctl, 0, 256, stream);
  hipMemsetAsync(out, 0, (size_t)T_TOK * D_MODEL * sizeof(float), stream);

  moe_cast_x_bf16<<<2048, 256, 0, stream>>>(x, xbf, T_TOK * D_MODEL / 4);
  moe_router<<<T_TOK / 8, 256, 0, stream>>>(x, rw, cnt, imp, loadc, topk_id, topk_w);
  moe_scan_aux<<<1, 32, 0, stream>>>(cnt, offs, cursor, imp, loadc,
                                     out + (size_t)T_TOK * D_MODEL);
  moe_scatter<<<T_TOK / 256, 256, 0, stream>>>(topk_id, topk_w, cursor, ptok, pgate);

  dim3 g1(D_FF / BN, MT_MAX, N_EXP);
  moe_ffn1<<<g1, 256, 0, stream>>>(xbf, w1, w3, offs, cnt, ptok, Hbuf);

  dim3 g2(D_MODEL / BN, MT_MAX, N_EXP);
  moe_ffn2<<<g2, 256, 0, stream>>>(Hbuf, w2, offs, cnt, ptok, pgate, out);
}
